// MiniGPT_89172110999737
// MI455X (gfx1250) — matrix operand order repaired and checked by reading, not compiled, not run on hardware
//
#include <hip/hip_runtime.h>
#include <hip/hip_bf16.h>
#include <math.h>

typedef _Float16 half_t;
typedef __attribute__((ext_vector_type(16))) _Float16 v16h;
typedef __attribute__((ext_vector_type(8)))  _Float16 v8h;
typedef __attribute__((ext_vector_type(4)))  _Float16 v4h;
typedef __attribute__((ext_vector_type(8)))  float    v8f;
typedef __attribute__((ext_vector_type(4)))  float    v4f;
typedef __attribute__((ext_vector_type(4)))  int      v4i;

#define D_MODEL 1024
#define NHEAD   16
#define HEADD   64
#define SEQLEN  1024
#define BATCH   2
#define NLAYER  6
#define FFN     4096
#define VOCAB   32000

// ---------------------------------------------------------------------------
// Async global->LDS copy (CDNA5 GLOBAL_LOAD_ASYNC_TO_LDS_B128, ASYNCcnt).
// Signature per clang diagnostic: (v4i AS1* gsrc, v4i AS3* ldst, Ii, Ii).
// Guarded: falls back to a register bounce if the builtin isn't declared.
// ---------------------------------------------------------------------------
#if defined(__AMDGCN__) && __has_builtin(__builtin_amdgcn_global_load_async_to_lds_b128)
#define HAS_ASYNC_LDS 1
typedef __attribute__((address_space(1))) v4i* as1_v4i_p;
typedef __attribute__((address_space(3))) v4i* as3_v4i_p;
__device__ inline void async_copy16(const half_t* g, half_t* l) {
  __builtin_amdgcn_global_load_async_to_lds_b128((as1_v4i_p)g, (as3_v4i_p)l, 0, 0);
}
__device__ inline void wait_async0() {
#if __has_builtin(__builtin_amdgcn_s_wait_asynccnt)
  __builtin_amdgcn_s_wait_asynccnt(0);
#else
  asm volatile("s_wait_asynccnt 0x0" ::: "memory");
#endif
}
#else
#define HAS_ASYNC_LDS 0
__device__ inline void async_copy16(const half_t* g, half_t* l) {
  *(v8h*)l = *(const v8h*)g;   // synchronous fallback (DScnt-ordered)
}
__device__ inline void wait_async0() {}
#endif

// ---------------------------------------------------------------------------
// WMMA fragment loaders (CDNA5 ISA 7.12.2 layouts, wave32)
// ---------------------------------------------------------------------------

// A matrix 16x32 f16: lane holds row (lane&15); group g=lane>>4:
// elements 0..7  -> K = g*8 + i ; elements 8..15 -> K = 16 + g*8 + (i-8)
__device__ inline v16h ld_a_frag(const half_t* base, int ld) {
  const int lane = threadIdx.x & 31;
  const int row  = lane & 15;
  const int g    = lane >> 4;
  const half_t* p = base + (size_t)row * ld;
  v8h lo = *(const v8h*)(p + g * 8);
  v8h hi = *(const v8h*)(p + 16 + g * 8);
  v16h r;
#pragma unroll
  for (int i = 0; i < 8; ++i) { r[i] = lo[i]; r[8 + i] = hi[i]; }
  return r;
}

// B matrix 32x16 f16 from a column-major LDS tile: element (k,n) at base[n*ld+k]
// lane holds col (lane&15); group g=lane>>4: elements 0..7 -> K = g*8 + i ; elements 8..15 -> K = 16 + g*8 + (i-8)
__device__ inline v16h ld_b_frag(const half_t* base, int ld) {
  const int lane = threadIdx.x & 31;
  const int col  = lane & 15;
  const int kb   = (lane >> 4) * 8;
  const half_t* p = base + (size_t)col * ld + kb;
  v8h lo = *(const v8h*)(p);
  v8h hi = *(const v8h*)(p + 16);
  v16h r;
#pragma unroll
  for (int i = 0; i < 8; ++i) { r[i] = lo[i]; r[8 + i] = hi[i]; }
  return r;
}

// ---------------------------------------------------------------------------
// fp32 -> f16 conversion (weights, once per call), vectorized b128 -> b64
// ---------------------------------------------------------------------------
__global__ void cvt_f16(const float* __restrict__ src, half_t* __restrict__ dst,
                        long long n4) {
  long long i = (long long)blockIdx.x * blockDim.x + threadIdx.x;
  long long stride = (long long)gridDim.x * blockDim.x;
  for (; i < n4; i += stride) {
    v4f v = ((const v4f*)src)[i];
    v4h h;
#pragma unroll
    for (int j = 0; j < 4; ++j) h[j] = (half_t)v[j];
    ((v4h*)dst)[i] = h;
  }
}

// ---------------------------------------------------------------------------
// embedding: x[b,s,:] = tok_emb[tokens[b,s],:] + pos_emb[s,:]
// ---------------------------------------------------------------------------
__global__ void embed_kernel(const int* __restrict__ tokens,
                             const float* __restrict__ tok_emb,
                             const float* __restrict__ pos_emb,
                             float* __restrict__ x) {
  int i   = blockIdx.x * blockDim.x + threadIdx.x;  // over BATCH*SEQ*D
  int d   = i & (D_MODEL - 1);
  int row = i >> 10;             // b*SEQ + s
  int s   = row & (SEQLEN - 1);
  int tok = tokens[row];
  x[(size_t)i] = tok_emb[(size_t)tok * D_MODEL + d] + pos_emb[(size_t)s * D_MODEL + d];
}

// ---------------------------------------------------------------------------
// LayerNorm over D=1024, one 256-thread block per row, f16 output
// ---------------------------------------------------------------------------
__global__ __launch_bounds__(256)
void layernorm_f16(const float* __restrict__ x, const float* __restrict__ scale,
                   const float* __restrict__ bias, half_t* __restrict__ y) {
  __shared__ float red[256];
  const int row = blockIdx.x;
  const int tid = threadIdx.x;
  const float* xr = x + (size_t)row * D_MODEL;
  float v[4];
  float s = 0.f;
#pragma unroll
  for (int i = 0; i < 4; ++i) { v[i] = xr[tid + i * 256]; s += v[i]; }
  red[tid] = s; __syncthreads();
  for (int st = 128; st > 0; st >>= 1) {
    if (tid < st) red[tid] += red[tid + st];
    __syncthreads();
  }
  const float mean = red[0] * (1.f / D_MODEL);
  __syncthreads();
  float s2 = 0.f;
#pragma unroll
  for (int i = 0; i < 4; ++i) { float d = v[i] - mean; s2 += d * d; }
  red[tid] = s2; __syncthreads();
  for (int st = 128; st > 0; st >>= 1) {
    if (tid < st) red[tid] += red[tid + st];
    __syncthreads();
  }
  const float var = red[0] * (1.f / D_MODEL);
  const float rs  = rsqrtf(var + 1e-5f);
#pragma unroll
  for (int i = 0; i < 4; ++i) {
    int c = tid + i * 256;
    y[(size_t)row * D_MODEL + c] = (half_t)((v[i] - mean) * rs * scale[c] + bias[c]);
  }
}

// ---------------------------------------------------------------------------
// WMMA GEMM: C[M,N] = A16[M,K] @ B16[K,N] (+ bias) (+ residual / GELU)
// 128x128 tile, BK=32, ping-pong double-buffered LDS:
//  - A tile staged via async global->LDS (no transpose needed)
//  - B tile fetched to registers early, transpose-stored to the *next* buffer
//    while WMMAs run on the current one
// EPI: 0 = bias->f16 (QKV); 1 = bias+residual->f32 (in-place safe);
//      2 = bias+exact GELU->f16 (MLP1); 3 = plain->f32 (LM head)
// ---------------------------------------------------------------------------
template <int EPI>
__global__ __launch_bounds__(256)
void gemm_wmma(const half_t* __restrict__ A, const half_t* __restrict__ B,
               const float* __restrict__ bias, const float* __restrict__ res,
               float* __restrict__ outF, half_t* __restrict__ outH,
               int M, int N, int K) {
  __shared__ half_t sA[2][128 * 32];  // row-major [m][k]
  __shared__ half_t sB[2][128 * 32];  // column-major [n][k]
  const int tid  = threadIdx.x;
  const int lane = tid & 31;
  const int w    = tid >> 5;
  const int m0   = blockIdx.y * 128;
  const int n0   = blockIdx.x * 128;
  const int wm   = (w & 3) * 32;   // wave sub-tile: 32 rows x 64 cols
  const int wn   = (w >> 2) * 64;

  v8f acc[2][4];
#pragma unroll
  for (int i = 0; i < 2; ++i)
#pragma unroll
    for (int j = 0; j < 4; ++j) acc[i][j] = {};

  const int arow = tid >> 1, acol = (tid & 1) * 16;  // A-tile loader coords
  const int bkr  = tid >> 3, bc0  = (tid & 7) * 16;  // B-tile loader coords

  // ---- prologue: stage tile k0=0 into buffer 0 ----
  {
    const half_t* gpA = A + (size_t)(m0 + arow) * K + acol;
    async_copy16(gpA,     sA[0] + arow * 32 + acol);
    async_copy16(gpA + 8, sA[0] + arow * 32 + acol + 8);
    const half_t* gpB = B + (size_t)bkr * N + n0 + bc0;
    v8h b0 = *(const v8h*)gpB;
    v8h b1 = *(const v8h*)(gpB + 8);
#pragma unroll
    for (int i = 0; i < 8; ++i) sB[0][(bc0 + i) * 32 + bkr] = b0[i];
#pragma unroll
    for (int i = 0; i < 8; ++i) sB[0][(bc0 + 8 + i) * 32 + bkr] = b1[i];
  }
  wait_async0();
  __syncthreads();

  int buf = 0;
  for (int k0 = 0; k0 < K; k0 += 32) {
    const bool more = (k0 + 32) < K;
    v8h b0 = {}, b1 = {};
    if (more) {  // kick off next tile: async A into buf^1, B into registers
      const half_t* gpA = A + (size_t)(m0 + arow) * K + (k0 + 32) + acol;
      async_copy16(gpA,     sA[buf ^ 1] + arow * 32 + acol);
      async_copy16(gpA + 8, sA[buf ^ 1] + arow * 32 + acol + 8);
      const half_t* gpB = B + (size_t)(k0 + 32 + bkr) * N + n0 + bc0;
      b0 = *(const v8h*)gpB;
      b1 = *(const v8h*)(gpB + 8);
    }

    // compute on current buffer
    v16h af[2], bf[4];
#pragma unroll
    for (int i = 0; i < 2; ++i) af[i] = ld_a_frag(sA[buf] + (wm + i * 16) * 32, 32);
#pragma unroll
    for (int j = 0; j < 4; ++j) bf[j] = ld_b_frag(sB[buf] + (wn + j * 16) * 32, 32);
#pragma unroll
    for (int i = 0; i < 2; ++i)
#pragma unroll
      for (int j = 0; j < 4; ++j)
        acc[i][j] = __builtin_amdgcn_wmma_f32_16x16x32_f16(
            false, af[i], false, bf[j], (short)0, acc[i][j], false, false);

    if (more) {  // transpose-store next B tile into the other buffer
#pragma unroll
      for (int i = 0; i < 8; ++i) sB[buf ^ 1][(bc0 + i) * 32 + bkr] = b0[i];
#pragma unroll
      for (int i = 0; i < 8; ++i) sB[buf ^ 1][(bc0 + 8 + i) * 32 + bkr] = b1[i];
    }
    wait_async0();
    __syncthreads();
    buf ^= 1;
  }

  const int g = lane >> 4;
#pragma unroll
  for (int i = 0; i < 2; ++i) {
#pragma unroll
    for (int j = 0; j < 4; ++j) {
      const int gc = n0 + wn + j * 16 + (lane & 15);
#pragma unroll
      for (int e = 0; e < 8; ++e) {
        const int gr = m0 + wm + i * 16 + e + 8 * g;
        float v = acc[i][j][e];
        if (EPI != 3) v += bias[gc];
        size_t o = (size_t)gr * N + gc;
        if (EPI == 0) {
          outH[o] = (half_t)v;
        } else if (EPI == 1) {
          outF[o] = res[o] + v;
        } else if (EPI == 2) {
          float gl = 0.5f * v * (1.f + erff(v * 0.70710678118f));
          outH[o] = (half_t)gl;
        } else {
          outF[o] = v;
        }
      }
    }
  }
}

// ---------------------------------------------------------------------------
// Flash-style causal attention. Block = (b, h, 128 q rows); 8 waves, each owns
// 16 q rows. K tile staged to LDS asynchronously (straight copy), V tile
// transpose-stored; online softmax with width-16 shuffles; P bounces through
// LDS to become a WMMA A fragment for P@V.
// q/k/v layout: f16 [b*SEQ, D_MODEL] with head h at column h*64.
// ---------------------------------------------------------------------------
__global__ __launch_bounds__(256)
void attn_kernel(const half_t* __restrict__ q16, const half_t* __restrict__ k16,
                 const half_t* __restrict__ v16, half_t* __restrict__ oh16) {
  __shared__ half_t sK[32 * 64];        // [key][d]
  __shared__ half_t sVt[64 * 32];       // [d][key]  (column-major B tile)
  __shared__ half_t sP[8 * 16 * 32];    // per-wave P tile [16][32]
  const int tid  = threadIdx.x;
  const int lane = tid & 31;
  const int w    = tid >> 5;
  const int g    = lane >> 4;
  const int q0   = blockIdx.x * 128;
  const int h    = blockIdx.y;
  const int b    = blockIdx.z;
  const size_t rowbase = (size_t)b * SEQLEN;

  // preload this wave's Q fragments (16 rows x 64 d = 2 A-frags)
  const int qrow = q0 + w * 16 + (lane & 15);
  v16h aq[2];
  {
    const half_t* qp = q16 + (rowbase + qrow) * D_MODEL + h * HEADD;
#pragma unroll
    for (int c = 0; c < 2; ++c) {
      v8h lo = *(const v8h*)(qp + c * 32 + g * 8);
      v8h hi = *(const v8h*)(qp + c * 32 + 16 + g * 8);
#pragma unroll
      for (int i = 0; i < 8; ++i) { aq[c][i] = lo[i]; aq[c][8 + i] = hi[i]; }
    }
  }

  v8f accO[4];
#pragma unroll
  for (int j = 0; j < 4; ++j) accO[j] = {};
  float m[8], l[8];
#pragma unroll
  for (int e = 0; e < 8; ++e) { m[e] = -1e30f; l[e] = 0.f; }

  half_t* sPw = sP + w * 16 * 32;
  const int kr = tid >> 3, d0 = (tid & 7) * 8;  // cooperative K/V loader coords
  const int kmax = q0 + 128;                    // causal: later tiles contribute 0

  for (int k0 = 0; k0 < kmax; k0 += 32) {
    {  // stage K (async copy) and V (transposed) tiles for this 32-key step
      const size_t gb = (rowbase + k0 + kr) * D_MODEL + h * HEADD + d0;
      async_copy16(k16 + gb, sK + kr * 64 + d0);
      v8h vv = *(const v8h*)(v16 + gb);
#pragma unroll
      for (int i = 0; i < 8; ++i) sVt[(d0 + i) * 32 + kr] = vv[i];
    }
    wait_async0();
    __syncthreads();

    // S = Q @ K^T for 16 q rows x 32 keys (two 16x16 accumulators)
    v8f s[2];
    s[0] = {}; s[1] = {};
#pragma unroll
    for (int hh = 0; hh < 2; ++hh)
#pragma unroll
      for (int c = 0; c < 2; ++c) {
        v16h bk = ld_b_frag(sK + (hh * 16) * 64 + c * 32, 64);
        s[hh] = __builtin_amdgcn_wmma_f32_16x16x32_f16(
            false, aq[c], false, bk, (short)0, s[hh], false, false);
      }

    // online softmax (rows live per half-wave: row = e + 8*g, col = lane&15)
#pragma unroll
    for (int e = 0; e < 8; ++e) {
      const int gq = q0 + w * 16 + e + 8 * g;
      float s0 = s[0][e] * 0.125f;  // 1/sqrt(64)
      float s1 = s[1][e] * 0.125f;
      if (k0 + (lane & 15) > gq)      s0 = -10000.f;
      if (k0 + 16 + (lane & 15) > gq) s1 = -10000.f;
      float mx = fmaxf(s0, s1);
#pragma unroll
      for (int off = 8; off >= 1; off >>= 1) mx = fmaxf(mx, __shfl_xor(mx, off, 16));
      const float nm = fmaxf(m[e], mx);
      const float sc = expf(m[e] - nm);
      const float p0 = expf(s0 - nm);
      const float p1 = expf(s1 - nm);
      float rs = p0 + p1;
#pragma unroll
      for (int off = 8; off >= 1; off >>= 1) rs += __shfl_xor(rs, off, 16);
      l[e] = l[e] * sc + rs;
      m[e] = nm;
#pragma unroll
      for (int j = 0; j < 4; ++j) accO[j][e] *= sc;
      sPw[(e + 8 * g) * 32 + (lane & 15)]      = (half_t)p0;
      sPw[(e + 8 * g) * 32 + 16 + (lane & 15)] = (half_t)p1;
    }
    __syncthreads();

    // O += P @ V   (P: 16x32 A-frag from LDS, V: four 32x16 B-frags)
    v16h ap = ld_a_frag(sPw, 32);
#pragma unroll
    for (int j = 0; j < 4; ++j) {
      v16h bv = ld_b_frag(sVt + (j * 16) * 32, 32);
      accO[j] = __builtin_amdgcn_wmma_f32_16x16x32_f16(
          false, ap, false, bv, (short)0, accO[j], false, false);
    }
    __syncthreads();
  }

  // normalize and write oh (f16, feeds the O-projection GEMM)
#pragma unroll
  for (int j = 0; j < 4; ++j) {
#pragma unroll
    for (int e = 0; e < 8; ++e) {
      const int gq = q0 + w * 16 + e + 8 * g;
      oh16[(rowbase + gq) * D_MODEL + h * HEADD + j * 16 + (lane & 15)] =
          (half_t)(accO[j][e] / l[e]);
    }
  }
}

// ---------------------------------------------------------------------------
// Host orchestration
// ---------------------------------------------------------------------------
extern "C" void kernel_launch(void* const* d_in, const int* in_sizes, int n_in,
                              void* d_out, int out_size, void* d_ws, size_t ws_size,
                              hipStream_t stream) {
  (void)in_sizes; (void)n_in; (void)out_size; (void)ws_size;
  const int*   tokens  = (const int*)d_in[0];
  const float* tok_emb = (const float*)d_in[1];
  const float* pos_emb = (const float*)d_in[2];
  const float* ln1_s   = (const float*)d_in[3];
  const float* ln1_b   = (const float*)d_in[4];
  const float* wq      = (const float*)d_in[5];
  const float* bq      = (const float*)d_in[6];
  const float* wk      = (const float*)d_in[7];
  const float* bk      = (const float*)d_in[8];
  const float* wv      = (const float*)d_in[9];
  const float* bv      = (const float*)d_in[10];
  const float* wo      = (const float*)d_in[11];
  const float* bo      = (const float*)d_in[12];
  const float* ln2_s   = (const float*)d_in[13];
  const float* ln2_b   = (const float*)d_in[14];
  const float* w1      = (const float*)d_in[15];
  const float* b1      = (const float*)d_in[16];
  const float* w2      = (const float*)d_in[17];
  const float* b2      = (const float*)d_in[18];
  const float* lnf_s   = (const float*)d_in[19];
  const float* lnf_b   = (const float*)d_in[20];
  const float* head_w  = (const float*)d_in[21];

  const int M = BATCH * SEQLEN;  // 2048 rows
  char* ws = (char*)d_ws;
  size_t off = 0;
  auto take = [&](size_t bytes) -> char* {
    char* p = ws + off;
    off += (bytes + 255) & ~(size_t)255;
    return p;
  };

  const size_t NW  = (size_t)NLAYER * D_MODEL * D_MODEL;  // per projection tensor
  const size_t NW1 = (size_t)NLAYER * D_MODEL * FFN;
  const size_t NWH = (size_t)D_MODEL * VOCAB;

  half_t* wq16   = (half_t*)take(NW  * 2);
  half_t* wk16   = (half_t*)take(NW  * 2);
  half_t* wv16   = (half_t*)take(NW  * 2);
  half_t* wo16   = (half_t*)take(NW  * 2);
  half_t* w116   = (half_t*)take(NW1 * 2);
  half_t* w216   = (half_t*)take(NW1 * 2);
  half_t* head16 = (half_t*)take(NWH * 2);
  float*  x      = (float*) take((size_t)M * D_MODEL * 4);
  half_t* y16    = (half_t*)take((size_t)M * D_MODEL * 2);
  half_t* q16    = (half_t*)take((size_t)M * D_MODEL * 2);
  half_t* k16    = (half_t*)take((size_t)M * D_MODEL * 2);
  half_t* v16    = (half_t*)take((size_t)M * D_MODEL * 2);
  half_t* oh16   = (half_t*)take((size_t)M * D_MODEL * 2);
  half_t* h16    = (half_t*)take((size_t)M * FFN * 2);

  // one-time (per call) weight conversion fp32 -> f16 (vectorized x4)
  cvt_f16<<<2048, 256, 0, stream>>>(wq, wq16, (long long)(NW / 4));
  cvt_f16<<<2048, 256, 0, stream>>>(wk, wk16, (long long)(NW / 4));
  cvt_f16<<<2048, 256, 0, stream>>>(wv, wv16, (long long)(NW / 4));
  cvt_f16<<<2048, 256, 0, stream>>>(wo, wo16, (long long)(NW / 4));
  cvt_f16<<<2048, 256, 0, stream>>>(w1, w116, (long long)(NW1 / 4));
  cvt_f16<<<2048, 256, 0, stream>>>(w2, w216, (long long)(NW1 / 4));
  cvt_f16<<<2048, 256, 0, stream>>>(head_w, head16, (long long)(NWH / 4));

  embed_kernel<<<(M * D_MODEL) / 256, 256, 0, stream>>>(tokens, tok_emb, pos_emb, x);

  const dim3 gProj(D_MODEL / 128, M / 128);  // (8,16)
  const dim3 gMlp1(FFN / 128, M / 128);      // (32,16)
  const dim3 gHead(VOCAB / 128, M / 128);    // (250,16)
  const dim3 gAttn(SEQLEN / 128, NHEAD, BATCH);

  for (int lyr = 0; lyr < NLAYER; ++lyr) {
    const size_t wofs = (size_t)lyr * D_MODEL * D_MODEL;
    const size_t fofs = (size_t)lyr * D_MODEL * FFN;

    layernorm_f16<<<M, 256, 0, stream>>>(x, ln1_s + lyr * D_MODEL, ln1_b + lyr * D_MODEL, y16);

    gemm_wmma<0><<<gProj, 256, 0, stream>>>(y16, wq16 + wofs, bq + lyr * D_MODEL,
                                            nullptr, nullptr, q16, M, D_MODEL, D_MODEL);
    gemm_wmma<0><<<gProj, 256, 0, stream>>>(y16, wk16 + wofs, bk + lyr * D_MODEL,
                                            nullptr, nullptr, k16, M, D_MODEL, D_MODEL);
    gemm_wmma<0><<<gProj, 256, 0, stream>>>(y16, wv16 + wofs, bv + lyr * D_MODEL,
                                            nullptr, nullptr, v16, M, D_MODEL, D_MODEL);

    attn_kernel<<<gAttn, 256, 0, stream>>>(q16, k16, v16, oh16);

    gemm_wmma<1><<<gProj, 256, 0, stream>>>(oh16, wo16 + wofs, bo + lyr * D_MODEL,
                                            x, x, nullptr, M, D_MODEL, D_MODEL);

    layernorm_f16<<<M, 256, 0, stream>>>(x, ln2_s + lyr * D_MODEL, ln2_b + lyr * D_MODEL, y16);

    gemm_wmma<2><<<gMlp1, 256, 0, stream>>>(y16, w116 + fofs, b1 + lyr * FFN,
                                            nullptr, nullptr, h16, M, FFN, D_MODEL);
    gemm_wmma<1><<<gProj, 256, 0, stream>>>(h16, w216 + fofs, b2 + lyr * D_MODEL,
                                            x, x, nullptr, M, D_MODEL, FFN);
  }

  layernorm_f16<<<M, 256, 0, stream>>>(x, lnf_s, lnf_b, y16);
  gemm_wmma<3><<<gHead, 256, 0, stream>>>(y16, head16, nullptr, nullptr,
                                          (float*)d_out, nullptr, M, VOCAB, D_MODEL);
}
